// ExtensibleAttention_69836168233357
// MI455X (gfx1250) — matrix operand order repaired and checked by reading, not compiled, not run on hardware
//
#include <hip/hip_runtime.h>
#include <hip/hip_bf16.h>
#include <math.h>

// ---------------------------------------------------------------------------
// Types for CDNA5 WMMA (wave32): v_wmma_f32_16x16x32_bf16
// ---------------------------------------------------------------------------
typedef __bf16 bf16_t;
typedef __attribute__((ext_vector_type(16))) __bf16 v16bf;   // 32 bytes
typedef __attribute__((ext_vector_type(8)))  __bf16 v8bf;    // 16 bytes
typedef __attribute__((ext_vector_type(8)))  float  v8f;

union V16U { v16bf v; v8bf h[2]; };

// ---------------------------------------------------------------------------
// Elementwise f32 -> bf16 (vectorized x4); n must be a multiple of 4.
// ---------------------------------------------------------------------------
__global__ __launch_bounds__(256)
void k_f32_to_bf16(const float* __restrict__ x, bf16_t* __restrict__ y, int n4) {
    int i = blockIdx.x * blockDim.x + threadIdx.x;
    if (i >= n4) return;
    float4 f = ((const float4*)x)[i];
    bf16_t* o = y + (size_t)i * 4;
    o[0] = (bf16_t)f.x; o[1] = (bf16_t)f.y; o[2] = (bf16_t)f.z; o[3] = (bf16_t)f.w;
}

// ---------------------------------------------------------------------------
// Weight transpose + convert: W [K,N] f32 row-major -> WT [N,K] bf16 row-major
// so B fragments become contiguous per-column K runs.
// ---------------------------------------------------------------------------
__global__ __launch_bounds__(256)
void k_transpose_bf16(const float* __restrict__ W, bf16_t* __restrict__ WT, int K, int N) {
    int t = blockIdx.x * blockDim.x + threadIdx.x;
    if (t >= K * N) return;
    int n = t / K;
    int k = t - n * K;
    WT[(size_t)n * K + k] = (bf16_t)W[(size_t)k * N + n];
}

// ---------------------------------------------------------------------------
// WMMA GEMM: out[M,Ntot] = A[M,Ktot](bf16) @ BT[Ntot,Ktot]^T(bf16) + bias
// One wave -> 16x64 output strip (4 WMMA tiles sharing the A fragment).
// Block = 256 threads = 8 waves = 128 rows; grid = (M/128, Ntot/64).
// ---------------------------------------------------------------------------
template <bool RELU, bool OUT_BF16>
__global__ __launch_bounds__(256)
void k_wmma_gemm(const bf16_t* __restrict__ A, const bf16_t* __restrict__ BT,
                 const float* __restrict__ bias, void* __restrict__ out,
                 int M, int Ntot, int Ktot) {
    const int wave = threadIdx.x >> 5;
    const int lane = threadIdx.x & 31;
    const int half = lane >> 4;      // 0: lanes 0-15, 1: lanes 16-31
    const int r    = lane & 15;
    const int tm   = blockIdx.x * 8 + wave;   // 16-row tile index
    const int n0   = blockIdx.y * 64;         // 64-column strip base
    if (tm * 16 >= M) return;

    const v8f vzero = {0.f, 0.f, 0.f, 0.f, 0.f, 0.f, 0.f, 0.f};
    v8f acc[4] = {vzero, vzero, vzero, vzero};

    // A fragment: lane r holds row tm*16+r; lanes 0-15 carry K 0-7 & 16-23,
    // lanes 16-31 carry K 8-15 & 24-31 (ISA 16-bit A 16x32 layout).
    const bf16_t* arow = A + (size_t)(tm * 16 + r) * Ktot;

    for (int k0 = 0; k0 < Ktot; k0 += 32) {
        V16U a;
        a.h[0] = *(const v8bf*)(arow + k0 + half * 8);         // K = kb .. kb+7
        a.h[1] = *(const v8bf*)(arow + k0 + 16 + half * 8);    // K = kb+16 .. kb+23
#pragma unroll
        for (int t = 0; t < 4; ++t) {
            // B fragment: lane r = column n0+t*16+r; same K split as A (0-7 & 16-23 / 8-15 & 24-31).
            const bf16_t* brow = BT + (size_t)(n0 + t * 16 + r) * Ktot + k0 + half * 8;
            V16U b;
            b.h[0] = *(const v8bf*)brow;
            b.h[1] = *(const v8bf*)(brow + 16);
            acc[t] = __builtin_amdgcn_wmma_f32_16x16x32_bf16(
                false, a.v, false, b.v, (short)0, acc[t], false, false);
        }
    }

    // C/D layout: VGPR i = row (half*8 + i), lane r = column.
    const int row0 = tm * 16 + half * 8;
#pragma unroll
    for (int t = 0; t < 4; ++t) {
        const int col = n0 + t * 16 + r;
        const float bv = bias ? bias[col] : 0.f;
#pragma unroll
        for (int i = 0; i < 8; ++i) {
            float x = acc[t][i] + bv;
            if (RELU) x = fmaxf(x, 0.f);
            const size_t idx = (size_t)(row0 + i) * Ntot + col;
            if (OUT_BF16) ((bf16_t*)out)[idx] = (bf16_t)x;
            else          ((float*)out)[idx]  = x;
        }
    }
}

// ---------------------------------------------------------------------------
// Fused attention elementwise: one thread per (n,l,h).
//   qh = q+pos, kh = k+pos ; qk = <qh,kh> over D=32
//   sp = ref + off ; w = relu(1-|sp.x-.5|)*relu(1-|sp.y-.5|)
//   attn = softmax(qk*w/sqrt(D)) over K=4 ; wv = sum(attn*w)
//   mid[:,h*32+d] = v*wv   (bf16 for final GEMM)
// ---------------------------------------------------------------------------
__global__ __launch_bounds__(256)
void k_attn(const float* __restrict__ qf, const float* __restrict__ kf,
            const float* __restrict__ vf, const float* __restrict__ posf,
            const float* __restrict__ offs, const float* __restrict__ refp,
            bf16_t* __restrict__ mid, int total_nl) {
    int t  = blockIdx.x * blockDim.x + threadIdx.x;
    int h  = t & 7;
    int nl = t >> 3;
    if (nl >= total_nl) return;

    const size_t base = (size_t)nl * 256 + (size_t)h * 32;
    const float4* q4 = (const float4*)(qf + base);
    const float4* k4 = (const float4*)(kf + base);
    const float4* p4 = (const float4*)(posf + base);

    float qk = 0.f;
#pragma unroll
    for (int j = 0; j < 8; ++j) {
        float4 q = q4[j], k = k4[j], p = p4[j];
        qk += (q.x + p.x) * (k.x + p.x);
        qk += (q.y + p.y) * (k.y + p.y);
        qk += (q.z + p.z) * (k.z + p.z);
        qk += (q.w + p.w) * (k.w + p.w);
    }

    const float* off = offs + (size_t)nl * 64 + (size_t)h * 8;  // [K=4][2]
    const float rx = refp[(size_t)nl * 2 + 0];
    const float ry = refp[(size_t)nl * 2 + 1];
    const float scale = 0.1767766952966369f;  // 1/sqrt(32)

    float w[4], logit[4], mx = -1e30f;
#pragma unroll
    for (int kk = 0; kk < 4; ++kk) {
        float cx = rx + off[kk * 2 + 0] - 0.5f;
        float cy = ry + off[kk * 2 + 1] - 0.5f;
        float wx = fmaxf(0.f, 1.f - fabsf(cx));
        float wy = fmaxf(0.f, 1.f - fabsf(cy));
        w[kk] = wx * wy;
        logit[kk] = qk * w[kk] * scale;
        mx = fmaxf(mx, logit[kk]);
    }
    float se = 0.f, wv = 0.f;
#pragma unroll
    for (int kk = 0; kk < 4; ++kk) {
        float e = __expf(logit[kk] - mx);
        se += e;
        wv += e * w[kk];
    }
    wv /= se;

    const float4* v4 = (const float4*)(vf + base);
    bf16_t* mo = mid + base;
#pragma unroll
    for (int j = 0; j < 8; ++j) {
        float4 v = v4[j];
        mo[j * 4 + 0] = (bf16_t)(v.x * wv);
        mo[j * 4 + 1] = (bf16_t)(v.y * wv);
        mo[j * 4 + 2] = (bf16_t)(v.z * wv);
        mo[j * 4 + 3] = (bf16_t)(v.w * wv);
    }
}

// ---------------------------------------------------------------------------
// Host-side orchestration (graph-capture safe: only kernel launches on stream)
// ---------------------------------------------------------------------------
extern "C" void kernel_launch(void* const* d_in, const int* in_sizes, int n_in,
                              void* d_out, int out_size, void* d_ws, size_t ws_size,
                              hipStream_t stream) {
    (void)n_in; (void)out_size; (void)ws_size;
    const int C = 256, C2 = 512, OFFW = 64;   // H*K*2 = 64
    const int M = in_sizes[0] / C;            // N*L = 65536

    const float* query = (const float*)d_in[0];
    const float* key_  = (const float*)d_in[1];
    const float* value = (const float*)d_in[2];
    const float* refp  = (const float*)d_in[3];
    const float* pose  = (const float*)d_in[4];
    const float* Wq  = (const float*)d_in[5];  const float* bq  = (const float*)d_in[6];
    const float* Wk  = (const float*)d_in[7];  const float* bk  = (const float*)d_in[8];
    const float* Wv  = (const float*)d_in[9];  const float* bv  = (const float*)d_in[10];
    const float* Wo1 = (const float*)d_in[11]; const float* bo1 = (const float*)d_in[12];
    const float* Wo2 = (const float*)d_in[13]; const float* bo2 = (const float*)d_in[14];
    const float* Wpos = (const float*)d_in[15]; const float* bpos = (const float*)d_in[16];
    const float* Wout = (const float*)d_in[17]; const float* bout = (const float*)d_in[18];

    // Workspace carve-out
    char* p = (char*)d_ws;
    auto alloc = [&](size_t bytes) -> void* {
        void* r = (void*)p;
        p += (bytes + 255) & ~(size_t)255;
        return r;
    };
    bf16_t* qbf   = (bf16_t*)alloc((size_t)M * C * 2);
    bf16_t* kbf   = (bf16_t*)alloc((size_t)M * C * 2);
    bf16_t* vbf   = (bf16_t*)alloc((size_t)M * C * 2);
    bf16_t* pbf   = (bf16_t*)alloc((size_t)M * C * 2);
    bf16_t* WqT   = (bf16_t*)alloc((size_t)C * C * 2);
    bf16_t* WkT   = (bf16_t*)alloc((size_t)C * C * 2);
    bf16_t* WvT   = (bf16_t*)alloc((size_t)C * C * 2);
    bf16_t* WposT = (bf16_t*)alloc((size_t)C * C * 2);
    bf16_t* Wo1T  = (bf16_t*)alloc((size_t)C2 * C * 2);
    bf16_t* Wo2T  = (bf16_t*)alloc((size_t)OFFW * C2 * 2);
    bf16_t* WoutT = (bf16_t*)alloc((size_t)C * C * 2);
    float*  qf    = (float*)alloc((size_t)M * C * 4);
    float*  kf    = (float*)alloc((size_t)M * C * 4);
    float*  vf    = (float*)alloc((size_t)M * C * 4);
    float*  posf  = (float*)alloc((size_t)M * C * 4);
    bf16_t* hid   = (bf16_t*)alloc((size_t)M * C2 * 2);
    float*  offs  = (float*)alloc((size_t)M * OFFW * 4);
    bf16_t* mid   = (bf16_t*)alloc((size_t)M * C * 2);

    // 1) activations f32 -> bf16
    {
        int n4 = (M * C) / 4;
        dim3 g((n4 + 255) / 256), b(256);
        k_f32_to_bf16<<<g, b, 0, stream>>>(query, qbf, n4);
        k_f32_to_bf16<<<g, b, 0, stream>>>(key_,  kbf, n4);
        k_f32_to_bf16<<<g, b, 0, stream>>>(value, vbf, n4);
        k_f32_to_bf16<<<g, b, 0, stream>>>(pose,  pbf, n4);
    }

    // 2) weights -> transposed bf16
    auto launch_tr = [&](const float* W, bf16_t* WT, int K, int N) {
        int n = K * N;
        k_transpose_bf16<<<dim3((n + 255) / 256), dim3(256), 0, stream>>>(W, WT, K, N);
    };
    launch_tr(Wq,   WqT,   C,  C);
    launch_tr(Wk,   WkT,   C,  C);
    launch_tr(Wv,   WvT,   C,  C);
    launch_tr(Wpos, WposT, C,  C);
    launch_tr(Wo1,  Wo1T,  C,  C2);
    launch_tr(Wo2,  Wo2T,  C2, OFFW);
    launch_tr(Wout, WoutT, C,  C);

    // 3) WMMA GEMMs
    const dim3 blk(256);
    auto gemm_grid = [&](int Ntot) { return dim3(M / 128, Ntot / 64); };
    // projections -> f32
    k_wmma_gemm<false, false><<<gemm_grid(C), blk, 0, stream>>>(qbf, WqT,   bq,   (void*)qf,   M, C,  C);
    k_wmma_gemm<false, false><<<gemm_grid(C), blk, 0, stream>>>(kbf, WkT,   bk,   (void*)kf,   M, C,  C);
    k_wmma_gemm<false, false><<<gemm_grid(C), blk, 0, stream>>>(vbf, WvT,   bv,   (void*)vf,   M, C,  C);
    k_wmma_gemm<false, false><<<gemm_grid(C), blk, 0, stream>>>(pbf, WposT, bpos, (void*)posf, M, C,  C);
    // offset MLP: relu(query@Wo1+bo1) -> bf16 hidden ; hidden@Wo2+bo2 -> f32 offs
    k_wmma_gemm<true,  true ><<<gemm_grid(C2), blk, 0, stream>>>(qbf, Wo1T, bo1, (void*)hid,  M, C2, C);
    k_wmma_gemm<false, false><<<gemm_grid(OFFW), blk, 0, stream>>>(hid, Wo2T, bo2, (void*)offs, M, OFFW, C2);

    // 4) fused attention elementwise -> bf16 mid
    {
        int total_threads = M * 8;  // H = 8
        k_attn<<<dim3(total_threads / 256), blk, 0, stream>>>(qf, kf, vf, posf, offs, refp, mid, M);
    }

    // 5) output projection straight into d_out (f32)
    k_wmma_gemm<false, false><<<gemm_grid(C), blk, 0, stream>>>(mid, WoutT, bout, d_out, M, C, C);
}
